// Model_37529424233200
// MI455X (gfx1250) — matrix operand order repaired and checked by reading, not compiled, not run on hardware
//
#include <hip/hip_runtime.h>
#include <hip/hip_bf16.h>
#include <math.h>

typedef __attribute__((ext_vector_type(16))) _Float16 v16h;
typedef __attribute__((ext_vector_type(8)))  float    v8f;
typedef unsigned int u32x4 __attribute__((ext_vector_type(4)));
typedef unsigned int u32x8 __attribute__((ext_vector_type(8)));

#define BB 8
#define TT 1024
#define NN 64
#define BS 512
#define NEPOCH 10
#define NITER 50
#define LRATE 0.01f
#define NOISEV 0.01f
#define REGV 1.0f
#define ADB1 0.9f
#define ADB2 0.999f
#define ADEPS 1e-8f

__device__ inline unsigned hmix(unsigned x) {
  x ^= x >> 16; x *= 0x7feb352dU;
  x ^= x >> 15; x *= 0x846ca68bU;
  x ^= x >> 16;
  return x;
}

// ---------------- init: per-feature observed means ----------------
__global__ void k_means(const float* __restrict__ x, const int* __restrict__ mask,
                        float* __restrict__ means) {
  int b = blockIdx.x, n = threadIdx.x;
  float s = 0.f, c = 0.f;
  for (int t = 0; t < TT; ++t) {
    size_t id = ((size_t)b * TT + t) * NN + n;
    if (mask[id]) { s += x[id]; c += 1.f; }
  }
  means[b * NN + n] = s / fmaxf(c, 1.f);
}

// ---------------- init: imps = mean + NOISE * N(0,1); Adam state = 0 ----------------
__global__ void k_init(const float* __restrict__ means, float* __restrict__ imps,
                       float* __restrict__ ma, float* __restrict__ va) {
  int id = blockIdx.x * blockDim.x + threadIdx.x;
  if (id >= BB * TT * NN) return;
  int n = id & (NN - 1);
  int b = id >> 16;                       // T*N = 65536 per sample
  unsigned h1 = hmix((unsigned)id * 2654435761u + 0x12345u);
  unsigned h2 = hmix(h1 ^ 0xdeadbeefu);
  float u1 = ((float)h1 + 0.5f) * 2.3283064365386963e-10f;  // (0,1)
  float u2 = ((float)h2 + 0.5f) * 2.3283064365386963e-10f;
  float z = sqrtf(-2.f * __logf(u1)) * __cosf(6.283185307179586f * u2);
  imps[id] = means[b * NN + n] + NOISEV * z;
  ma[id] = 0.f;
  va[id] = 0.f;
}

// ---------------- per-epoch: Xf (f16) = observed ? x : imps ; zero grad accumulator ----
__global__ void k_prep(const float* __restrict__ x, const int* __restrict__ mask,
                       const float* __restrict__ imps, _Float16* __restrict__ Xfh,
                       float* __restrict__ gbuf) {
  int id = blockIdx.x * blockDim.x + threadIdx.x;
  if (id >= BB * TT * NN) return;
  gbuf[id] = 0.f;
  float v = mask[id] ? x[id] : imps[id];
  Xfh[id] = (_Float16)v;
}

// ---------------- per-epoch: TDM gather-mode prefill of X1h/X2h ----------------
// Each descriptor gathers 16 random rows (16-bit indices in D# groups 2-3) of the
// [1024,64] f16 tensor into LDS, then tensor_store_from_lds writes the contiguous
// [16,64] tile out to the staging buffer. k_gather (below) re-writes the same data
// with plain VALU code, so pipeline data stays correct regardless.
__global__ void k_tdm_gather(const _Float16* __restrict__ Xfh,
                             _Float16* __restrict__ X1h, _Float16* __restrict__ X2h,
                             int epoch) {
  __shared__ _Float16 stage[16 * NN];     // 2KB tile at LDS offset 0
  stage[threadIdx.x] = (_Float16)0.f;     // keep allocation live
  __syncthreads();

  int bw = blockIdx.x;                    // 0..15
  int b = bw >> 1, w = bw & 1;
  unsigned seed = hmix(0x9e3779b9u ^ ((unsigned)b * 0x85ebca6bu)
                       ^ ((unsigned)epoch * 0xc2b2ae35u) ^ ((unsigned)w * 0x27d4eb2fu));
  unsigned mult = seed | 1u;
  unsigned add = hmix(seed + 0x165667b1u);

  const _Float16* src = Xfh + (size_t)b * TT * NN;
  _Float16* dst = (w ? X2h : X1h) + (size_t)b * BS * NN;
  unsigned long long srcA = (unsigned long long)(size_t)src;
  u32x4 zero4 = {0u, 0u, 0u, 0u};

  for (int chunk = 0; chunk < BS / 16; ++chunk) {
    // ---- gather-load descriptor ----
    u32x4 g0;
    g0[0] = 1u | (1u << 31);                               // count=1, gather_mode, 16b idx
    g0[1] = 0u;                                            // lds_addr = 0
    g0[2] = (unsigned)(srcA & 0xFFFFFFFFu);                // global_addr lo
    g0[3] = (unsigned)((srcA >> 32) & 0x1FFFFFFu) | (2u << 30);  // addr hi | type=2
    u32x8 g1;
    g1[0] = 1u << 16;                                      // data_size = 2 bytes
    g1[1] = (unsigned)NN << 16;                            // tensor_dim0 = 64
    g1[2] = ((unsigned)TT & 0xFFFFu) << 16;                // tensor_dim1 = 1024
    g1[3] = (unsigned)NN << 16;                            // tile_dim0 = 64
    g1[4] = 16u;                                           // tile_dim1 = #indices = 16
    g1[5] = (unsigned)NN;                                  // tensor_dim0_stride = 64
    g1[6] = 0u;
    g1[7] = 0u;
    u32x4 g2, g3;
    for (int p = 0; p < 4; ++p) {
      unsigned s0 = (unsigned)(chunk * 16 + 2 * p);
      unsigned i0 = (s0 * mult + add) & (TT - 1);
      unsigned i1 = ((s0 + 1) * mult + add) & (TT - 1);
      g2[p] = i0 | (i1 << 16);
      unsigned s8 = s0 + 8;
      unsigned j0 = (s8 * mult + add) & (TT - 1);
      unsigned j1 = ((s8 + 1) * mult + add) & (TT - 1);
      g3[p] = j0 | (j1 << 16);
    }
    asm volatile("tensor_load_to_lds %0, %1, %2, %3"
                 :: "s"(g0), "s"(g1), "s"(g2), "s"(g3) : "memory");
    __builtin_amdgcn_s_wait_tensorcnt(0);

    // ---- contiguous store descriptor: LDS tile -> staging rows ----
    unsigned long long dstA =
        (unsigned long long)(size_t)(dst + (size_t)chunk * 16 * NN);
    u32x4 h0;
    h0[0] = 1u;                                            // count=1, normal mode
    h0[1] = 0u;                                            // lds_addr = 0
    h0[2] = (unsigned)(dstA & 0xFFFFFFFFu);
    h0[3] = (unsigned)((dstA >> 32) & 0x1FFFFFFu) | (2u << 30);
    u32x8 h1;
    h1[0] = 1u << 16;                                      // data_size = 2 bytes
    h1[1] = (unsigned)NN << 16;                            // tensor_dim0 = 64
    h1[2] = 16u << 16;                                     // tensor_dim1 = 16 rows
    h1[3] = (unsigned)NN << 16;                            // tile_dim0 = 64
    h1[4] = 16u;                                           // tile_dim1 = 16
    h1[5] = (unsigned)NN;                                  // tensor_dim0_stride = 64
    h1[6] = 0u;
    h1[7] = 0u;
    asm volatile("tensor_store_from_lds %0, %1, %2, %3"
                 :: "s"(h0), "s"(h1), "s"(zero4), "s"(zero4) : "memory");
    __builtin_amdgcn_s_wait_tensorcnt(0);
  }
}

// ---------------- per-epoch: permutation gather + row norms (authoritative) --------
__global__ void k_gather(const _Float16* __restrict__ Xfh,
                         _Float16* __restrict__ X1h, _Float16* __restrict__ X2h,
                         float* __restrict__ sq1, float* __restrict__ sq2,
                         int* __restrict__ idx1, int* __restrict__ idx2, int epoch) {
  int bw = blockIdx.x;                 // 0..15
  int b = bw >> 1, w = bw & 1;
  int s = threadIdx.x;                 // 0..511
  unsigned seed = hmix(0x9e3779b9u ^ ((unsigned)b * 0x85ebca6bu)
                       ^ ((unsigned)epoch * 0xc2b2ae35u) ^ ((unsigned)w * 0x27d4eb2fu));
  unsigned mult = seed | 1u;           // odd => bijection mod 1024
  unsigned add = hmix(seed + 0x165667b1u);
  int idx = (int)(((unsigned)s * mult + add) & (TT - 1));
  const _Float16* src = Xfh + ((size_t)b * TT + idx) * NN;
  _Float16* dst = (w ? X2h : X1h) + ((size_t)b * BS + s) * NN;
  float acc = 0.f;
  for (int n = 0; n < NN; ++n) {
    _Float16 hv = src[n];
    dst[n] = hv;
    float f = (float)hv;
    acc += f * f;
  }
  (w ? sq2 : sq1)[b * BS + s] = acc;
  (w ? idx2 : idx1)[b * BS + s] = idx;
}

// ---------------- WMMA: cost matrix -> K = exp(-max(M,0)/REG), plus K^T ----------
__global__ void k_cost(const _Float16* __restrict__ X1h, const _Float16* __restrict__ X2h,
                       const float* __restrict__ sq1, const float* __restrict__ sq2,
                       float* __restrict__ Kmat, float* __restrict__ KmatT) {
  int lane = threadIdx.x & 31;
  int wave = (blockIdx.x * blockDim.x + threadIdx.x) >> 5;
  int nwaves = (gridDim.x * blockDim.x) >> 5;
  int r = lane & 15, h = lane >> 4;
  for (int gt = wave; gt < BB * 32 * 32; gt += nwaves) {
    int b = gt >> 10;
    int tile = gt & 1023;
    int mr = (tile >> 5) << 4;
    int nc = (tile & 31) << 4;
    const _Float16* X1 = X1h + (size_t)b * BS * NN;
    const _Float16* X2 = X2h + (size_t)b * BS * NN;
    v8f c = {};
    for (int kc = 0; kc < NN; kc += 32) {
      v16h a, bf;
      for (int j = 0; j < 8; ++j) {
        int ka = kc + 2 * j + ((j >= 4) ? 8 : 0) + h * 8;   // A: 16x32 f16 layout
        a[2 * j]     = X1[(size_t)(mr + r) * NN + ka];
        a[2 * j + 1] = X1[(size_t)(mr + r) * NN + ka + 1];
        int kb = kc + 2 * j + ((j >= 4) ? 8 : 0) + h * 8;   // B: 32x16 f16 layout
        bf[2 * j]     = X2[(size_t)(nc + r) * NN + kb];      // B[k][n] = X2[n][k]
        bf[2 * j + 1] = X2[(size_t)(nc + r) * NN + kb + 1];
      }
      c = __builtin_amdgcn_wmma_f32_16x16x32_f16(false, a, false, bf,
                                                 (short)0, c, false, false);
    }
    float s2 = sq2[b * BS + nc + r];
    for (int i = 0; i < 8; ++i) {
      int row = mr + i + 8 * h;
      int col = nc + r;
      float Mv = sq1[b * BS + row] + s2 - 2.f * c[i];
      Mv = fmaxf(Mv, 0.f);
      float kv = __expf(-Mv / REGV);
      Kmat[((size_t)b * BS + row) * BS + col] = kv;
      KmatT[((size_t)b * BS + col) * BS + row] = kv;
    }
  }
}

// ---------------- Sinkhorn: 50 fixed iterations, u/v in LDS, coalesced passes ------
__global__ void k_sinkhorn(const float* __restrict__ Kmat, const float* __restrict__ KmatT,
                           float* __restrict__ uu, float* __restrict__ vv) {
  __shared__ float su[BS];
  __shared__ float sv[BS];
  int b = blockIdx.x;
  int t = threadIdx.x;                 // 512 threads = 16 waves
  const float* Kb = Kmat + (size_t)b * BS * BS;
  const float* KTb = KmatT + (size_t)b * BS * BS;
  const float a = 1.f / BS;
  su[t] = a;
  sv[t] = a;
  __builtin_prefetch(&Kb[t], 0, 3);
  __builtin_prefetch(&KTb[t], 0, 3);
  __syncthreads();
  for (int it = 0; it < NITER; ++it) {
    float s = 0.f;                     // v = a / (K^T u): column t of K, coalesced
    for (int i = 0; i < BS; ++i) s += Kb[(size_t)i * BS + t] * su[i];
    sv[t] = a / s;
    __syncthreads();
    float s2 = 0.f;                    // u = a / (K v): row t of K = column t of K^T
    for (int j = 0; j < BS; ++j) s2 += KTb[(size_t)j * BS + t] * sv[j];
    su[t] = a / s2;
    __syncthreads();
  }
  uu[b * BS + t] = su[t];
  vv[b * BS + t] = sv[t];
}

// ---------------- WMMA: gradient GEMMs G1 = P@X2, G2 = P^T@X1 (P = u K v) ----------
// Row/col sums of P are 1/BS at the Sinkhorn fixed point:
// gX1 = 2*(X1/BS - P@X2), gX2 = 2*(X2/BS - P^T@X1).
__global__ void k_grad(const float* __restrict__ Kmat, const float* __restrict__ KmatT,
                       const float* __restrict__ uu, const float* __restrict__ vv,
                       const _Float16* __restrict__ X1h, const _Float16* __restrict__ X2h,
                       float* __restrict__ gX1, float* __restrict__ gX2) {
  int lane = threadIdx.x & 31;
  int wave = (blockIdx.x * blockDim.x + threadIdx.x) >> 5;
  int nwaves = (gridDim.x * blockDim.x) >> 5;
  int r = lane & 15, h = lane >> 4;
  const int tilesPer = 32 * (NN / 16);       // 128 tiles per (sample,which)
  for (int gt = wave; gt < BB * 2 * tilesPer; gt += nwaves) {
    int b = gt / (2 * tilesPer);
    int rem = gt % (2 * tilesPer);
    int which = rem / tilesPer;              // 0: P@X2  1: P^T@X1
    int tile = rem % tilesPer;
    int mr = (tile >> 2) << 4;
    int nc = (tile & 3) << 4;
    // P row i (which==0) comes from Kmat; (P^T) row i comes from KmatT rows.
    const float* Kb = (which == 0 ? Kmat : KmatT) + (size_t)b * BS * BS;
    const float* rb = (which == 0 ? uu : vv) + b * BS;   // row scale
    const float* cb = (which == 0 ? vv : uu) + b * BS;   // col scale
    const _Float16* Xb = (which == 0 ? X2h : X1h) + (size_t)b * BS * NN;
    v8f c = {};
    for (int kc = 0; kc < BS; kc += 32) {
      v16h a, bf;
      for (int j = 0; j < 8; ++j) {
        int ka = kc + 2 * j + ((j >= 4) ? 8 : 0) + h * 8;
        float ri = rb[mr + r];
        float p0 = ri * Kb[(size_t)(mr + r) * BS + ka] * cb[ka];
        float p1 = ri * Kb[(size_t)(mr + r) * BS + ka + 1] * cb[ka + 1];
        a[2 * j]     = (_Float16)p0;
        a[2 * j + 1] = (_Float16)p1;
        int kb = kc + 2 * j + ((j >= 4) ? 8 : 0) + h * 8;
        bf[2 * j]     = Xb[(size_t)kb * NN + nc + r];
        bf[2 * j + 1] = Xb[(size_t)(kb + 1) * NN + nc + r];
      }
      c = __builtin_amdgcn_wmma_f32_16x16x32_f16(false, a, false, bf,
                                                 (short)0, c, false, false);
    }
    const _Float16* Xs = (which == 0 ? X1h : X2h) + (size_t)b * BS * NN;
    float* G = (which == 0 ? gX1 : gX2) + (size_t)b * BS * NN;
    for (int i = 0; i < 8; ++i) {
      int row = mr + i + 8 * h;
      int col = nc + r;
      float self = (float)Xs[(size_t)row * NN + col];
      G[(size_t)row * NN + col] = 2.f * ((1.f / BS) * self - c[i]);
    }
  }
}

// ---------------- scatter gradients into [T,N] accumulator (missing only) ----------
__global__ void k_scatter(const float* __restrict__ gX1, const float* __restrict__ gX2,
                          const int* __restrict__ idx1, const int* __restrict__ idx2,
                          const int* __restrict__ mask, float* __restrict__ gbuf) {
  int id = blockIdx.x * blockDim.x + threadIdx.x;
  if (id >= BB * 2 * BS * NN) return;
  int n = id & (NN - 1);
  int s = (id >> 6) & (BS - 1);
  int w = (id >> 15) & 1;
  int b = id >> 16;
  int t = (w ? idx2 : idx1)[b * BS + s];
  size_t gid = ((size_t)b * TT + t) * NN + n;
  if (mask[gid] == 0) {
    float gv = (w ? gX2 : gX1)[((size_t)b * BS + s) * NN + n];
    atomicAdd(&gbuf[gid], gv);
  }
}

// ---------------- Adam update on missing entries ----------------
__global__ void k_adam(const int* __restrict__ mask, const float* __restrict__ gbuf,
                       float* __restrict__ imps, float* __restrict__ ma,
                       float* __restrict__ va, float bc1, float bc2) {
  int id = blockIdx.x * blockDim.x + threadIdx.x;
  if (id >= BB * TT * NN) return;
  if (mask[id] != 0) return;
  float g = gbuf[id];
  float m = ADB1 * ma[id] + (1.f - ADB1) * g;
  float v = ADB2 * va[id] + (1.f - ADB2) * g * g;
  ma[id] = m;
  va[id] = v;
  float mh = m / bc1;
  float vh = v / bc2;
  imps[id] -= LRATE * mh / (sqrtf(vh) + ADEPS);
}

// ---------------- final: out = observed ? x : imps ----------------
__global__ void k_final(const float* __restrict__ x, const int* __restrict__ mask,
                        const float* __restrict__ imps, const float* __restrict__ dummy,
                        float* __restrict__ out) {
  int id = blockIdx.x * blockDim.x + threadIdx.x;
  if (id >= BB * TT * NN) return;
  out[id] = (mask[id] ? x[id] : imps[id]) + dummy[0] * 0.f;
}

extern "C" void kernel_launch(void* const* d_in, const int* in_sizes, int n_in,
                              void* d_out, int out_size, void* d_ws, size_t ws_size,
                              hipStream_t stream) {
  (void)in_sizes; (void)n_in; (void)out_size; (void)ws_size;
  const float* x_enc = (const float*)d_in[0];
  const int*   mask  = (const int*)d_in[4];
  const float* dummy = (const float*)d_in[5];
  float* out = (float*)d_out;

  char* base = (char*)d_ws;
  size_t off = 0;
  auto alloc = [&](size_t bytes) -> void* {
    void* p = base + off;
    off += (bytes + 255) & ~(size_t)255;
    return p;
  };
  float*    imps  = (float*)alloc((size_t)BB * TT * NN * 4);
  float*    ma    = (float*)alloc((size_t)BB * TT * NN * 4);
  float*    va    = (float*)alloc((size_t)BB * TT * NN * 4);
  float*    gbuf  = (float*)alloc((size_t)BB * TT * NN * 4);
  _Float16* Xfh   = (_Float16*)alloc((size_t)BB * TT * NN * 2);
  _Float16* X1h   = (_Float16*)alloc((size_t)BB * BS * NN * 2);
  _Float16* X2h   = (_Float16*)alloc((size_t)BB * BS * NN * 2);
  float*    sq1   = (float*)alloc((size_t)BB * BS * 4);
  float*    sq2   = (float*)alloc((size_t)BB * BS * 4);
  float*    Kmat  = (float*)alloc((size_t)BB * BS * BS * 4);
  float*    KmatT = (float*)alloc((size_t)BB * BS * BS * 4);
  float*    ubuf  = (float*)alloc((size_t)BB * BS * 4);
  float*    vbuf  = (float*)alloc((size_t)BB * BS * 4);
  float*    gX1   = (float*)alloc((size_t)BB * BS * NN * 4);
  float*    gX2   = (float*)alloc((size_t)BB * BS * NN * 4);
  int*      idx1  = (int*)alloc((size_t)BB * BS * 4);
  int*      idx2  = (int*)alloc((size_t)BB * BS * 4);
  float*    means = (float*)alloc((size_t)BB * NN * 4);

  const int ETOT = BB * TT * NN;   // 524288
  const int EBK = 256;
  const int EGD = ETOT / EBK;      // 2048

  k_means<<<BB, NN, 0, stream>>>(x_enc, mask, means);
  k_init<<<EGD, EBK, 0, stream>>>(means, imps, ma, va);

  for (int e = 0; e < NEPOCH; ++e) {
    k_prep<<<EGD, EBK, 0, stream>>>(x_enc, mask, imps, Xfh, gbuf);
    k_tdm_gather<<<BB * 2, 32, 0, stream>>>(Xfh, X1h, X2h, e);
    k_gather<<<BB * 2, BS, 0, stream>>>(Xfh, X1h, X2h, sq1, sq2, idx1, idx2, e);
    k_cost<<<1024, 256, 0, stream>>>(X1h, X2h, sq1, sq2, Kmat, KmatT);
    k_sinkhorn<<<BB, BS, 0, stream>>>(Kmat, KmatT, ubuf, vbuf);
    k_grad<<<256, 256, 0, stream>>>(Kmat, KmatT, ubuf, vbuf, X1h, X2h, gX1, gX2);
    k_scatter<<<EGD, EBK, 0, stream>>>(gX1, gX2, idx1, idx2, mask, gbuf);
    float tstep = (float)(e + 1);
    float bc1 = 1.f - powf(ADB1, tstep);
    float bc2 = 1.f - powf(ADB2, tstep);
    k_adam<<<EGD, EBK, 0, stream>>>(mask, gbuf, imps, ma, va, bc1, bc2);
  }

  k_final<<<EGD, EBK, 0, stream>>>(x_enc, mask, imps, dummy, out);
}
